// RingAdvancedDistributedDilatedAttention_8615704396429
// MI455X (gfx1250) — matrix operand order repaired and checked by reading, not compiled, not run on hardware
//
#include <hip/hip_runtime.h>
#include <hip/hip_bf16.h>

typedef __attribute__((ext_vector_type(16))) _Float16 v16h;
typedef __attribute__((ext_vector_type(8)))  float    v8f;
typedef unsigned int v4u __attribute__((ext_vector_type(4)));
typedef int          v4i __attribute__((ext_vector_type(4)));
typedef int          v8i __attribute__((ext_vector_type(8)));

#define E_DIM   768
#define NSEQ    8192
#define HD      64
#define QKVN    2304
#define LN_EPS  1e-5f

#if defined(__has_builtin)
#if __has_builtin(__builtin_amdgcn_tensor_load_to_lds) && __has_builtin(__builtin_amdgcn_s_wait_tensorcnt)
#define USE_TDM 1
#endif
#endif

// ---------- helpers ----------
__device__ __forceinline__ v16h ld_frag(const _Float16* p0, const _Float16* p1) {
    union { v16h h; uint4 u[2]; } f;
    f.u[0] = *reinterpret_cast<const uint4*>(p0);
    f.u[1] = *reinterpret_cast<const uint4*>(p1);
    return f.h;
}

__device__ __forceinline__ v8f wmma16(v16h a, v16h b, v8f c) {
    return __builtin_amdgcn_wmma_f32_16x16x32_f16(
        /*neg_a=*/false, a, /*neg_b=*/false, b,
        /*c_mod=*/(short)0, c, /*reuse_a=*/false, /*reuse_b=*/false);
}

// ---------- precision conversion ----------
__global__ void k_cvt_f32_f16(const float* __restrict__ in, _Float16* __restrict__ out, int n) {
    int i = blockIdx.x * blockDim.x + threadIdx.x;
    int stride = gridDim.x * blockDim.x;
    for (; i < n; i += stride) out[i] = (_Float16)in[i];
}

__global__ void k_zero_f32(float* __restrict__ p, int n) {
    int i = blockIdx.x * blockDim.x + threadIdx.x;
    int stride = gridDim.x * blockDim.x;
    for (; i < n; i += stride) p[i] = 0.0f;
}

// ---------- shared GEMM mainloop: C(128x64 per WG) = A(MxK) * W(NxK)^T ----------
// A row-major K-contiguous, W row-major K-contiguous. 4 waves, wave w does a
// 32x64 tile (rows mbase+32w..+31): 2 A-fragments, 8 accumulators, 8 WMMAs per
// 32-k step against an LDS-staged 64x32 W tile shared by the whole workgroup.
__device__ __forceinline__ void gemm_tile2(const _Float16* __restrict__ A,
                                           const _Float16* __restrict__ W,
                                           int K, int mbase, int nbase,
                                           _Float16* Bs, v8f acc[2][4]) {
    const int lane = threadIdx.x & 31;
    const int wave = threadIdx.x >> 5;
    const int l2 = lane & 15, hi = lane >> 4;
    const int t  = threadIdx.x;
    const int sn = t >> 1, sk = (t & 1) * 16;
    const int arow0 = mbase + wave * 32 + l2;
    for (int r = 0; r < 2; ++r)
        for (int j = 0; j < 4; ++j)
            for (int i = 0; i < 8; ++i) acc[r][j][i] = 0.0f;

    for (int kk = 0; kk < K; kk += 32) {
        __syncthreads();
        const _Float16* src = W + (size_t)(nbase + sn) * K + kk + sk;
        *reinterpret_cast<uint4*>(Bs + sn * 32 + sk)     = *reinterpret_cast<const uint4*>(src);
        *reinterpret_cast<uint4*>(Bs + sn * 32 + sk + 8) = *reinterpret_cast<const uint4*>(src + 8);
        __syncthreads();
        const _Float16* ap0 = A + (size_t)arow0 * K + kk;
        const _Float16* ap1 = ap0 + (size_t)16 * K;
        v16h a0 = ld_frag(ap0 + hi * 16, ap0 + hi * 16 + 8);
        v16h a1 = ld_frag(ap1 + hi * 16, ap1 + hi * 16 + 8);
#pragma unroll
        for (int j = 0; j < 4; ++j) {
            const _Float16* bp = Bs + (j * 16 + l2) * 32 + hi * 16;
            v16h b = ld_frag(bp, bp + 8);
            acc[0][j] = wmma16(a0, b, acc[0][j]);
            acc[1][j] = wmma16(a1, b, acc[1][j]);
        }
    }
}

// ---------- QKV projection: writes f16 q/k/v, split by column ----------
__global__ void __launch_bounds__(128)
k_gemm_qkv(const _Float16* __restrict__ x16, const _Float16* __restrict__ w16,
           const float* __restrict__ bias,
           _Float16* __restrict__ q16, _Float16* __restrict__ k16, _Float16* __restrict__ v16) {
    __shared__ __align__(16) _Float16 Bs[64 * 32];
    const int nbase = blockIdx.x * 64, mbase = blockIdx.y * 128;
    v8f acc[2][4];
    gemm_tile2(x16, w16, E_DIM, mbase, nbase, Bs, acc);
    const int lane = threadIdx.x & 31, wave = threadIdx.x >> 5;
    const int l2 = lane & 15, hi = lane >> 4;
#pragma unroll
    for (int j = 0; j < 4; ++j) {
        const int col = nbase + j * 16 + l2;
        const float bv = bias[col];
        _Float16* dst; int c2;
        if (col < E_DIM)          { dst = q16; c2 = col; }
        else if (col < 2 * E_DIM) { dst = k16; c2 = col - E_DIM; }
        else                      { dst = v16; c2 = col - 2 * E_DIM; }
#pragma unroll
        for (int r = 0; r < 2; ++r)
#pragma unroll
            for (int v = 0; v < 8; ++v) {
                const int row = mbase + wave * 32 + r * 16 + v + 8 * hi;
                dst[(size_t)row * E_DIM + c2] = (_Float16)(acc[r][j][v] + bv);
            }
    }
}

// ---------- flash-attention over one (group, head, segment) block ----------
// All groups reduce to 2048x2048 attention with d=64 after dilation.
__global__ void __launch_bounds__(128)
k_attn(const _Float16* __restrict__ q16, const _Float16* __restrict__ k16,
       const _Float16* __restrict__ v16, float* __restrict__ o32) {
    __shared__ __align__(16) _Float16 Ks[32 * 64];     // key tile, row-major [key][d]
    __shared__ __align__(16) _Float16 Vt[64 * 40];     // V transposed [d][key], stride 40
    __shared__ __align__(16) _Float16 Ps[4][16 * 32];  // per-wave P staging

    const int ab = blockIdx.y;
    int head, seg, slen, dil, off;
    if (ab < 16)      { head = ab >> 2;            seg = ab & 3;  slen = 2048; dil = 1; off = 0; }
    else if (ab < 24) { int u = ab - 16; head = 4 + (u >> 1); seg = u & 1; slen = 4096; dil = 2; off = 1; }
    else              { head = 8 + (ab - 24);      seg = 0;       slen = 8192; dil = 4; off = 2; }

    const int lane = threadIdx.x & 31, wave = threadIdx.x >> 5;
    const int l2 = lane & 15, hi = lane >> 4;
    const int qt0 = blockIdx.x * 64 + wave * 16;   // dilated-row base for this wave
    const float scale = 0.125f;                    // 64^-0.5

    // Q fragments live in registers for the whole KV sweep
    const size_t qpos = (size_t)(seg * slen + off + (qt0 + l2) * dil);
    const _Float16* qp = q16 + qpos * E_DIM + head * HD;
    const v16h aq0 = ld_frag(qp + hi * 16,      qp + 8 + hi * 16);
    const v16h aq1 = ld_frag(qp + 32 + hi * 16, qp + 40 + hi * 16);

    v8f m_run, l_run, oacc[4];
    for (int i = 0; i < 8; ++i) { m_run[i] = -3.0e38f; l_run[i] = 0.0f; }
    for (int j = 0; j < 4; ++j)
        for (int i = 0; i < 8; ++i) oacc[j][i] = 0.0f;

    const int t = threadIdx.x;
    const int sr = t >> 2, sc = (t & 3) * 16;      // staging row / d-chunk

#ifdef USE_TDM
    // Loop-invariant D# group 1: 2D tile 64(halves) x 32(rows), data_size=2B,
    // row stride = dil*E_DIM elements (the dilation gather expressed as a stride).
    v8i g1;
    g1[0] = (1 << 16);           // workgroup_mask=0, data_size=1 (2 bytes)
    g1[1] = (64 << 16);          // tensor_dim0[15:0]=64 in bits 63:48
    g1[2] = (32 << 16);          // tensor_dim0 hi=0 | tensor_dim1[15:0]=32
    g1[3] = (64 << 16);          // tensor_dim1 hi=0 | tile_dim0=64
    g1[4] = 32;                  // tile_dim1=32, tile_dim2=0
    g1[5] = dil * E_DIM;         // tensor_dim0_stride[31:0]
    g1[6] = 0;
    g1[7] = 0;
    v4i z4; z4[0] = 0; z4[1] = 0; z4[2] = 0; z4[3] = 0;
#if __clang_major__ >= 23
    v8i z8; for (int i = 0; i < 8; ++i) z8[i] = 0;
#endif
    const unsigned lds_ks = (unsigned)(size_t)(void*)Ks;
#endif

    for (int kt = 0; kt < 64; ++kt) {
        __syncthreads();
        const size_t kpos = (size_t)(seg * slen + off + (kt * 32 + sr) * dil);
#ifdef USE_TDM
        if (wave == 0) {
            const unsigned long long ga = (unsigned long long)(size_t)
                (k16 + (size_t)(seg * slen + off + kt * 32 * dil) * E_DIM + head * HD);
            v4u g0;
            g0[0] = 1u;                                   // count=1, user mode
            g0[1] = lds_ks;                               // lds_addr
            g0[2] = (unsigned)(ga & 0xffffffffu);         // global_addr[31:0]
            g0[3] = (unsigned)((ga >> 32) & 0x01ffffffu)  // global_addr[56:32]
                    | (2u << 30);                         // type=2 ("image")
#if __clang_major__ >= 23
            __builtin_amdgcn_tensor_load_to_lds(g0, g1, z4, z4, z8, 0);
#else
            __builtin_amdgcn_tensor_load_to_lds(g0, g1, z4, z4, 0);
#endif
            __builtin_amdgcn_s_wait_tensorcnt(0);
        }
#else
        const _Float16* kp = k16 + kpos * E_DIM + head * HD + sc;
        *reinterpret_cast<uint4*>(Ks + sr * 64 + sc)     = *reinterpret_cast<const uint4*>(kp);
        *reinterpret_cast<uint4*>(Ks + sr * 64 + sc + 8) = *reinterpret_cast<const uint4*>(kp + 8);
#endif
        // V staged transposed so P*V B-fragments are contiguous in LDS
        union { uint4 u[2]; _Float16 h[16]; } vb;
        const _Float16* vp = v16 + kpos * E_DIM + head * HD + sc;
        vb.u[0] = *reinterpret_cast<const uint4*>(vp);
        vb.u[1] = *reinterpret_cast<const uint4*>(vp + 8);
#pragma unroll
        for (int e = 0; e < 16; ++e) Vt[(sc + e) * 40 + sr] = vb.h[e];
        if (kt + 1 < 64) {  // hint next V tile toward the caches
            const _Float16* nv = v16 + (size_t)(seg * slen + off + ((kt + 1) * 32 + sr) * dil) * E_DIM;
            __builtin_prefetch(nv, 0, 1);
        }
        __syncthreads();

        // S = scale * Q K^T   (two 16x16 column tiles, K-dim 64 = 2 WMMAs each)
        v8f s0, s1;
        for (int i = 0; i < 8; ++i) { s0[i] = 0.0f; s1[i] = 0.0f; }
        {
            const _Float16* b = Ks + l2 * 64 + hi * 16;
            s0 = wmma16(aq0, ld_frag(b, b + 8), s0);
            b = Ks + l2 * 64 + 32 + hi * 16;
            s0 = wmma16(aq1, ld_frag(b, b + 8), s0);
            b = Ks + (16 + l2) * 64 + hi * 16;
            s1 = wmma16(aq0, ld_frag(b, b + 8), s1);
            b = Ks + (16 + l2) * 64 + 32 + hi * 16;
            s1 = wmma16(aq1, ld_frag(b, b + 8), s1);
        }
        for (int i = 0; i < 8; ++i) { s0[i] *= scale; s1[i] *= scale; }

        // streaming softmax: rows live across each 16-lane half in C layout
        v8f tmax;
        for (int i = 0; i < 8; ++i) tmax[i] = fmaxf(s0[i], s1[i]);
        for (int d = 1; d < 16; d <<= 1)
            for (int i = 0; i < 8; ++i) tmax[i] = fmaxf(tmax[i], __shfl_xor(tmax[i], d));
        v8f m_new, alpha, p0, p1, psum;
        for (int i = 0; i < 8; ++i) {
            m_new[i] = fmaxf(m_run[i], tmax[i]);
            alpha[i] = __expf(m_run[i] - m_new[i]);
            p0[i] = __expf(s0[i] - m_new[i]);
            p1[i] = __expf(s1[i] - m_new[i]);
            psum[i] = p0[i] + p1[i];
        }
        for (int d = 1; d < 16; d <<= 1)
            for (int i = 0; i < 8; ++i) psum[i] += __shfl_xor(psum[i], d);
        for (int i = 0; i < 8; ++i) {
            l_run[i] = l_run[i] * alpha[i] + psum[i];
            m_run[i] = m_new[i];
        }
        for (int j = 0; j < 4; ++j)
            for (int i = 0; i < 8; ++i) oacc[j][i] *= alpha[i];

        // C-layout -> A-layout for P via per-wave LDS round trip
        _Float16* pw = Ps[wave];
#pragma unroll
        for (int i = 0; i < 8; ++i) {
            const int prow = i + 8 * hi;
            pw[prow * 32 + l2]      = (_Float16)p0[i];
            pw[prow * 32 + 16 + l2] = (_Float16)p1[i];
        }
        __syncthreads();
        const v16h ap = ld_frag(pw + l2 * 32 + hi * 16, pw + l2 * 32 + 8 + hi * 16);
#pragma unroll
        for (int j = 0; j < 4; ++j) {
            const _Float16* bv = Vt + (j * 16 + l2) * 40 + hi * 16;
            oacc[j] = wmma16(ap, ld_frag(bv, bv + 8), oacc[j]);
        }
    }

    // O = acc / l, scattered to dilated positions
    v8f rl;
    for (int i = 0; i < 8; ++i) rl[i] = 1.0f / l_run[i];
#pragma unroll
    for (int j = 0; j < 4; ++j)
        for (int i = 0; i < 8; ++i) {
            const int row_t = qt0 + i + 8 * hi;
            const size_t opos = (size_t)(seg * slen + off + row_t * dil);
            o32[opos * E_DIM + head * HD + j * 16 + l2] = oacc[j][i] * rl[i];
        }
}

// ---------- LayerNorm (MAGNETO sub-LN) + f16 convert ----------
__global__ void __launch_bounds__(256)
k_ln(const float* __restrict__ o32, const float* __restrict__ gamma,
     const float* __restrict__ beta, _Float16* __restrict__ on16) {
    const int row = blockIdx.x;
    const float* src = o32 + (size_t)row * E_DIM;
    float v[3];
    float s = 0.0f;
#pragma unroll
    for (int i = 0; i < 3; ++i) { v[i] = src[threadIdx.x + i * 256]; s += v[i]; }
    __shared__ float red[8];
    for (int d = 1; d < 32; d <<= 1) s += __shfl_xor(s, d);
    const int wave = threadIdx.x >> 5, lane = threadIdx.x & 31;
    if (lane == 0) red[wave] = s;
    __syncthreads();
    float tot = 0.0f;
    for (int w = 0; w < 8; ++w) tot += red[w];
    const float mu = tot * (1.0f / E_DIM);
    float s2 = 0.0f;
#pragma unroll
    for (int i = 0; i < 3; ++i) { const float d0 = v[i] - mu; s2 += d0 * d0; }
    for (int d = 1; d < 32; d <<= 1) s2 += __shfl_xor(s2, d);
    __syncthreads();
    if (lane == 0) red[wave] = s2;
    __syncthreads();
    float tv = 0.0f;
    for (int w = 0; w < 8; ++w) tv += red[w];
    const float rstd = rsqrtf(tv * (1.0f / E_DIM) + LN_EPS);
#pragma unroll
    for (int i = 0; i < 3; ++i) {
        const int c = threadIdx.x + i * 256;
        on16[(size_t)row * E_DIM + c] = (_Float16)((v[i] - mu) * rstd * gamma[c] + beta[c]);
    }
}

// ---------- output projection, fp32 result ----------
__global__ void __launch_bounds__(128)
k_gemm_out(const _Float16* __restrict__ on16, const _Float16* __restrict__ w16,
           const float* __restrict__ bias, float* __restrict__ out) {
    __shared__ __align__(16) _Float16 Bs[64 * 32];
    const int nbase = blockIdx.x * 64, mbase = blockIdx.y * 128;
    v8f acc[2][4];
    gemm_tile2(on16, w16, E_DIM, mbase, nbase, Bs, acc);
    const int lane = threadIdx.x & 31, wave = threadIdx.x >> 5;
    const int l2 = lane & 15, hi = lane >> 4;
#pragma unroll
    for (int j = 0; j < 4; ++j) {
        const int col = nbase + j * 16 + l2;
        const float bv = bias[col];
#pragma unroll
        for (int r = 0; r < 2; ++r)
#pragma unroll
            for (int v = 0; v < 8; ++v) {
                const int row = mbase + wave * 32 + r * 16 + v + 8 * hi;
                out[(size_t)row * E_DIM + col] = acc[r][j][v] + bv;
            }
    }
}

// ---------- host orchestration ----------
extern "C" void kernel_launch(void* const* d_in, const int* in_sizes, int n_in,
                              void* d_out, int out_size, void* d_ws, size_t ws_size,
                              hipStream_t stream) {
    const float* x       = (const float*)d_in[0];  // [8192, 768]
    const float* w_qkv   = (const float*)d_in[1];  // [2304, 768]
    const float* b_qkv   = (const float*)d_in[2];  // [2304]
    const float* w_out   = (const float*)d_in[3];  // [768, 768]
    const float* b_out   = (const float*)d_in[4];  // [768]
    const float* ln_g    = (const float*)d_in[5];  // [768]
    const float* ln_b    = (const float*)d_in[6];  // [768]
    float* out = (float*)d_out;                    // [8192, 768]

    char* ws = (char*)d_ws;
    size_t off = 0;
    auto take = [&](size_t bytes) { char* p = ws + off; off += (bytes + 255) & ~(size_t)255; return p; };
    _Float16* x16  = (_Float16*)take((size_t)NSEQ * E_DIM * 2);
    _Float16* wq16 = (_Float16*)take((size_t)QKVN * E_DIM * 2);
    _Float16* wo16 = (_Float16*)take((size_t)E_DIM * E_DIM * 2);
    _Float16* q16  = (_Float16*)take((size_t)NSEQ * E_DIM * 2);
    _Float16* k16  = (_Float16*)take((size_t)NSEQ * E_DIM * 2);
    _Float16* v16  = (_Float16*)take((size_t)NSEQ * E_DIM * 2);
    float*    o32  = (float*)   take((size_t)NSEQ * E_DIM * 4);
    _Float16* on16 = (_Float16*)take((size_t)NSEQ * E_DIM * 2);
    (void)ws_size; (void)in_sizes; (void)n_in; (void)out_size;

    // 1) fp32 -> f16 conversions
    k_cvt_f32_f16<<<2048, 256, 0, stream>>>(x, x16, NSEQ * E_DIM);
    k_cvt_f32_f16<<<1024, 256, 0, stream>>>(w_qkv, wq16, QKVN * E_DIM);
    k_cvt_f32_f16<<<512, 256, 0, stream>>>(w_out, wo16, E_DIM * E_DIM);

    // 2) QKV projection (WMMA GEMM), split into q/k/v f16
    k_gemm_qkv<<<dim3(QKVN / 64, NSEQ / 128), 128, 0, stream>>>(x16, wq16, b_qkv, q16, k16, v16);

    // 3) zero attention output (dilated groups leave gaps that must be 0)
    k_zero_f32<<<2048, 256, 0, stream>>>(o32, NSEQ * E_DIM);

    // 4) dilated flash attention: 28 blocks x 32 query tiles of 64 rows
    k_attn<<<dim3(32, 28), 128, 0, stream>>>(q16, k16, v16, o32);

    // 5) MAGNETO LayerNorm + f16 convert
    k_ln<<<NSEQ, 256, 0, stream>>>(o32, ln_g, ln_b, on16);

    // 6) output projection, fp32
    k_gemm_out<<<dim3(E_DIM / 64, NSEQ / 128), 128, 0, stream>>>(on16, wo16, b_out, out);
}
